// Net_77592879169659
// MI455X (gfx1250) — matrix operand order repaired and checked by reading, not compiled, not run on hardware
//
#include <hip/hip_runtime.h>
#include <hip/hip_bf16.h>
#include <math.h>

// ---------------- problem constants (from reference) ----------------
#define N_NODES 25000
#define N_EDGES 100000
#define FEAT    32
#define DIM     64
#define NGRAPH  125
#define HID     128
#define NPG     (N_NODES / NGRAPH)      // 200 nodes per graph (batch = repeat(arange(G), N//G))
#define KTOT    (HID * DIM + DIM)       // 8256: (h,i) pairs + 64 rows for b2
#define KBLK    (KTOT / 32)             // 258 K-blocks of 32
static_assert(N_EDGES % 16 == 0, "edge tiles assumed full");
static_assert(KTOT % 32 == 0, "K must be multiple of 32");

// TDM chunking of the reorganized W stream
#define CHUNK        6                       // K-blocks per TDM chunk
#define NCHUNK       (KBLK / CHUNK)          // 43
static_assert(KBLK % CHUNK == 0, "chunking must tile K exactly");
#define KB_ELEMS     (4 * 32 * 16)           // bf16 elements per K-block (one B strip)
#define CHUNK_ELEMS  (CHUNK * KB_ELEMS)      // 12288 bf16 = 24 KB
#define CHUNK_BYTES  (CHUNK_ELEMS * 2)
#define WT_ELEMS     ((size_t)KBLK * KB_ELEMS)
#define WT_QWORDS    ((unsigned)(WT_ELEMS / 4))  // 8-byte elements in Wt = 132096

typedef __attribute__((ext_vector_type(16))) __bf16       v16bf;
typedef __attribute__((ext_vector_type(8)))  float        v8f;
typedef __attribute__((ext_vector_type(4)))  unsigned int v4u;
typedef __attribute__((ext_vector_type(4)))  int          v4i;
typedef __attribute__((ext_vector_type(8)))  int          v8i;

__device__ __forceinline__ void atomic_add_f32(float* p, float v) {
  __hip_atomic_fetch_add(p, v, __ATOMIC_RELAXED, __HIP_MEMORY_SCOPE_AGENT);
}
__device__ __forceinline__ float sigmoidf(float x) { return 1.0f / (1.0f + expf(-x)); }

// ---------------------------------------------------------------------------
// TDM: DMA one 24KB chunk of the W stream into LDS (tracked by TENSORcnt).
// D# per CDNA5 ISA: group0 = {count=1, lds_addr,
// global_addr[56:0], type=2}; group1 = {data_size=8B, tensor_dim0 = whole
// stream (no OOB), tensor_dim1=1, tile_dim0 = 3072 x 8B, tile_dim1=1,
// dim0_stride = tensor_dim0}; groups 2/3 zero (<=2D tensor).
// ---------------------------------------------------------------------------
__device__ __forceinline__ void tdm_load_chunk(const __bf16* gsrc, unsigned lds_off) {
  unsigned long long ga = (unsigned long long)(size_t)gsrc;
  v4u g0;
  g0[0] = 1u;                                            // count=1, no gather
  g0[1] = lds_off;                                       // LDS byte address
  g0[2] = (unsigned)ga;                                  // global_addr[31:0]
  g0[3] = (unsigned)((ga >> 32) & 0x01FFFFFFu)           // global_addr[56:32]
          | 0x80000000u;                                 // type=2 ("image")
  v8i g1;
  g1[0] = (3 << 16);                                     // data_size = 8 bytes
  g1[1] = (int)((WT_QWORDS & 0xFFFFu) << 16);            // tensor_dim0[15:0]
  g1[2] = (int)((WT_QWORDS >> 16) | (1u << 16));         // tensor_dim0[31:16] | tensor_dim1=1
  g1[3] = (int)((unsigned)(CHUNK_BYTES / 8) << 16);      // tile_dim0 = 3072
  g1[4] = 1;                                             // tile_dim1 = 1, tile_dim2 = 0
  g1[5] = (int)WT_QWORDS;                                // tensor_dim0_stride[31:0]
  g1[6] = 0;                                             // stride[47:32] | dim1_stride lo
  g1[7] = 0;
  v4i z4 = {0, 0, 0, 0};
#if defined(__clang_major__) && (__clang_major__ >= 23)
  v8i z8 = {0, 0, 0, 0, 0, 0, 0, 0};
  __builtin_amdgcn_tensor_load_to_lds(g0, g1, z4, z4, z8, 0);
#else
  __builtin_amdgcn_tensor_load_to_lds(g0, g1, z4, z4, 0);
#endif
}

// ---------------------------------------------------------------------------
// W2 reorganization: w2[128,4096] (+ b2[4096]) -> Wt bf16, WMMA-B layout:
// element = ((kb*4 + cb)*32 + lane)*16 + t, lane = g*16 + n, K = kb*32+g*16+t,
// K<8192: w2[h*4096+i*64+col] (h=K/64, i=K%64); else b2[(K-8192)*64+col].
// ---------------------------------------------------------------------------
__global__ __launch_bounds__(256) void prep_w_kernel(const float* __restrict__ w2,
                                                     const float* __restrict__ b2,
                                                     __bf16* __restrict__ Wt) {
  int idx = blockIdx.x * blockDim.x + threadIdx.x;
  if (idx >= (int)WT_ELEMS) return;
  int t    = idx & 15;
  int lane = (idx >> 4) & 31;
  int cb   = (idx >> 9) & 3;
  int kb   = idx >> 11;
  int g = lane >> 4, n = lane & 15;
  int K   = kb * 32 + g * 16 + t;
  int col = cb * 16 + n;
  float v;
  if (K < HID * DIM) {
    int h = K >> 6, i = K & 63;
    v = w2[(size_t)h * (DIM * DIM) + i * DIM + col];
  } else {
    int i = K - HID * DIM;
    v = b2[i * DIM + col];
  }
  Wt[idx] = (__bf16)v;
}

// ---------------------------------------------------------------------------
// Fused NNConv edge-message GEMM (105 GFLOP / head):
//   msg[16-edge tile, 64] = P @ W, P[e,(h,i)] = relu(ea@w1+b1)[e,h]*x[src_e,i]
// One wave per 16-edge tile, NW tiles per block. The shared W stream is
// TDM-staged into double-buffered LDS by wave 0 (TENSORcnt + barriers) and
// consumed by all waves; A is built on the fly; f32 atomic scatter to targets.
// ---------------------------------------------------------------------------
#define NW 4
__global__ __launch_bounds__(NW * 32) void edge_msg_wmma(
    const float* __restrict__ xfeat,      // [N,64]
    const int*   __restrict__ src,
    const int*   __restrict__ tgt,
    const float* __restrict__ edge_attr,  // [E,5]
    const float* __restrict__ w1,         // [5,128]
    const float* __restrict__ b1,         // [128]
    const __bf16* __restrict__ Wt,        // [KBLK][4][32][16] bf16 (prep_w layout)
    float* __restrict__ accum,            // [N,64] f32, atomically accumulated
    int E) {
  __shared__ float  sx [NW][16][DIM];     // x[src] tile (f32)          16 KB
  __shared__ __bf16 shE[NW][16][HID];     // edge-MLP hidden (relu)     16 KB
  __shared__ int    ssrc[NW][16];
  __shared__ int    stgt[NW][16];
  __shared__ __attribute__((aligned(128))) __bf16 sB[2][CHUNK_ELEMS];  // 48 KB

  const int wave = threadIdx.x >> 5;
  const int lane = threadIdx.x & 31;
  const int tile = blockIdx.x * NW + wave;
  const bool active = (tile * 16) < E;    // wave-uniform; wave 0 is always active

  // LDS byte offsets of the two B buffers (flat addr low 32 bits = LDS offset)
  const unsigned ldsB0 = (unsigned)(size_t)(const void*)&sB[0][0];
  const unsigned ldsB1 = (unsigned)(size_t)(const void*)&sB[1][0];

  // kick off chunk 0 DMA; it overlaps the sx/shE staging below
  if (wave == 0) tdm_load_chunk(Wt, ldsB0);

  if (active && lane < 16) {
    int e = tile * 16 + lane;
    ssrc[wave][lane] = src[e];
    stgt[wave][lane] = tgt[e];
  }
  __syncthreads();

  if (active) {
    // stage x[src] rows: 16x64 f32, 32 values per lane
    for (int t = 0; t < 32; ++t) {
      int idx = t * 32 + lane;            // 0..1023
      int row = idx >> 6, col = idx & 63;
      sx[wave][row][col] = xfeat[(size_t)ssrc[wave][row] * DIM + col];
    }
    // edge-MLP layer 1 (5 MACs/value): each lane -> one edge, 64 h values
    int e = lane & 15;
    int hbase = (lane >> 4) * 64;
    size_t eg = (size_t)(tile * 16 + e) * 5;
    float a0 = edge_attr[eg + 0], a1 = edge_attr[eg + 1], a2 = edge_attr[eg + 2];
    float a3 = edge_attr[eg + 3], a4 = edge_attr[eg + 4];
    for (int hh = 0; hh < 64; ++hh) {
      int h = hbase + hh;
      float acc = b1[h] + a0 * w1[0 * HID + h] + a1 * w1[1 * HID + h] +
                  a2 * w1[2 * HID + h] + a3 * w1[3 * HID + h] + a4 * w1[4 * HID + h];
      shE[wave][e][h] = (__bf16)fmaxf(acc, 0.0f);
    }
  }
  __syncthreads();

  const int m  = lane & 15;               // A-matrix row = edge-in-tile
  const int g8 = (lane >> 4) * 8;         // half the A K base for lane group (k = 2*g8 + t)
  v8f acc[4] = {v8f{}, v8f{}, v8f{}, v8f{}};

  for (int c = 0; c < NCHUNK; ++c) {
    if (wave == 0) {
      if (c + 1 < NCHUNK) {
        // prefetch next chunk into the other buffer, then wait for current
        tdm_load_chunk(Wt + (size_t)(c + 1) * CHUNK_ELEMS,
                       ((c + 1) & 1) ? ldsB1 : ldsB0);
        __builtin_amdgcn_s_wait_tensorcnt(1);   // chunk c complete, c+1 in flight
      } else {
        __builtin_amdgcn_s_wait_tensorcnt(0);   // final chunk complete
      }
    }
    __syncthreads();                            // sB[c&1] valid for all waves

    if (active) {
      const __bf16* bbase = &sB[c & 1][0];
#pragma unroll
      for (int kk = 0; kk < CHUNK; ++kk) {
        int kb = c * CHUNK + kk;
        // within an aligned 32-wide K block, h is constant, i spans 32 values
        float coef; int i0;
        if (kb < 2 * HID) { coef = (float)shE[wave][m][kb >> 1]; i0 = (kb & 1) << 5; }
        else              { coef = 1.0f;                         i0 = (kb - 2 * HID) << 5; }
        v16bf a;
#pragma unroll
        for (int t = 0; t < 16; ++t) {
          int kl = 2 * g8 + t;   // contiguous K: k = 16*(lane/16) + t, matches B
          a[t] = (__bf16)(coef * sx[wave][m][i0 + kl]);
        }
        const __bf16* bp = bbase + kk * KB_ELEMS + lane * 16;
#pragma unroll
        for (int cb = 0; cb < 4; ++cb) {
          v16bf b = *(const v16bf*)(bp + cb * (32 * 16));
          acc[cb] = __builtin_amdgcn_wmma_f32_16x16x32_bf16(
              false, a, false, b, (short)0, acc[cb], false, false);
        }
      }
    }
    __syncthreads();                            // done reading sB[c&1] before reuse
  }

  if (active) {
    // scatter: C/D layout lane L: n = L%16, rows M = v + 8*(L/16)
    const int gg = lane >> 4, n = lane & 15;
#pragma unroll
    for (int cb = 0; cb < 4; ++cb)
#pragma unroll
      for (int v = 0; v < 8; ++v) {
        int node = stgt[wave][v + gg * 8];
        atomic_add_f32(accum + (size_t)node * DIM + cb * 16 + n, acc[cb][v]);
      }
  }
}

// ---------------------------- small VALU kernels ----------------------------
__global__ __launch_bounds__(256) void deg_kernel(const int* __restrict__ tgt,
                                                  float* __restrict__ deg, int E) {
  int e = blockIdx.x * blockDim.x + threadIdx.x;
  if (e < E) atomic_add_f32(deg + tgt[e], 1.0f);
}

__global__ __launch_bounds__(256) void lin0_kernel(const float* __restrict__ x,
                                                   const float* __restrict__ w,   // [32,64]
                                                   const float* __restrict__ b,
                                                   float* __restrict__ out, int N) {
  int idx = blockIdx.x * blockDim.x + threadIdx.x;
  if (idx >= N * DIM) return;
  int n = idx >> 6, o = idx & 63;
  const float* xr = x + (size_t)n * FEAT;
  float acc = b[o];
  for (int k = 0; k < FEAT; ++k) acc += xr[k] * w[k * DIM + o];
  out[idx] = fmaxf(acc, 0.0f);
}

// accum -> mean + bias + relu   (dst may alias accum)
__global__ __launch_bounds__(256) void node_update(const float* __restrict__ accum,
                                                   const float* __restrict__ deg,
                                                   const float* __restrict__ bias,
                                                   float* __restrict__ dst, int N) {
  int idx = blockIdx.x * blockDim.x + threadIdx.x;
  if (idx >= N * DIM) return;
  int n = idx >> 6, o = idx & 63;
  float d = fmaxf(deg[n], 1.0f);
  dst[idx] = fmaxf(accum[idx] / d + bias[o], 0.0f);
}

__global__ __launch_bounds__(256) void gru_kernel(const float* __restrict__ m,
                                                  const float* __restrict__ h,
                                                  const float* __restrict__ w_ih,  // [192,64]
                                                  const float* __restrict__ b_ih,
                                                  const float* __restrict__ w_hh,  // [192,64]
                                                  const float* __restrict__ b_hh,
                                                  float* __restrict__ hout, int N) {
  int idx = blockIdx.x * blockDim.x + threadIdx.x;
  if (idx >= N * DIM) return;
  int n = idx >> 6, d = idx & 63;
  const float* mr = m + (size_t)n * DIM;
  const float* hr = h + (size_t)n * DIM;
  float rx = b_ih[d], zx = b_ih[DIM + d], nx = b_ih[2 * DIM + d];
  float rh = b_hh[d], zh = b_hh[DIM + d], nh = b_hh[2 * DIM + d];
  for (int j = 0; j < DIM; ++j) {
    float mj = mr[j], hj = hr[j];
    rx += mj * w_ih[(size_t)d * DIM + j];
    zx += mj * w_ih[(size_t)(DIM + d) * DIM + j];
    nx += mj * w_ih[(size_t)(2 * DIM + d) * DIM + j];
    rh += hj * w_hh[(size_t)d * DIM + j];
    zh += hj * w_hh[(size_t)(DIM + d) * DIM + j];
    nh += hj * w_hh[(size_t)(2 * DIM + d) * DIM + j];
  }
  float r  = sigmoidf(rx + rh);
  float z  = sigmoidf(zx + zh);
  float nn = tanhf(nx + r * nh);
  hout[idx] = (1.0f - z) * nn + z * hr[d];
}

// ------------------------------- set2set ------------------------------------
__global__ __launch_bounds__(64) void s2s_lstm(const float* __restrict__ qstar, // [G,128]
                                               float* __restrict__ h,           // [G,64]
                                               float* __restrict__ c,           // [G,64]
                                               const float* __restrict__ w_ih,  // [256,128]
                                               const float* __restrict__ b_ih,
                                               const float* __restrict__ w_hh,  // [256,64]
                                               const float* __restrict__ b_hh) {
  int g = blockIdx.x, d = threadIdx.x;
  __shared__ float sq[2 * DIM];
  __shared__ float shh[DIM];
  shh[d] = h[(size_t)g * DIM + d];
  sq[d] = qstar[(size_t)g * 2 * DIM + d];
  sq[d + DIM] = qstar[(size_t)g * 2 * DIM + DIM + d];
  __syncthreads();
  float gate[4];
  for (int k = 0; k < 4; ++k) {
    int row = k * DIM + d;
    float acc = b_ih[row] + b_hh[row];
    for (int j = 0; j < 2 * DIM; ++j) acc += sq[j] * w_ih[(size_t)row * 2 * DIM + j];
    for (int j = 0; j < DIM; ++j)     acc += shh[j] * w_hh[(size_t)row * DIM + j];
    gate[k] = acc;
  }
  float i = sigmoidf(gate[0]), f = sigmoidf(gate[1]);
  float gg = tanhf(gate[2]),   o = sigmoidf(gate[3]);
  float cn = f * c[(size_t)g * DIM + d] + i * gg;
  float hn = o * tanhf(cn);
  c[(size_t)g * DIM + d] = cn;
  h[(size_t)g * DIM + d] = hn;
}

// softmax-attention + readout over the fixed 200 nodes of graph g; qstar=[h, r]
__global__ __launch_bounds__(256) void s2s_attn(const float* __restrict__ xn,  // [N,64]
                                                const float* __restrict__ h,   // [G,64]
                                                float* __restrict__ qstar) {   // [G,128]
  int g = blockIdx.x, tid = threadIdx.x;
  __shared__ float sh[DIM];
  __shared__ float se[256];
  __shared__ float sa[NPG];
  if (tid < DIM) sh[tid] = h[(size_t)g * DIM + tid];
  __syncthreads();
  const size_t base = (size_t)g * NPG;
  float lmax = -INFINITY;
  for (int i = tid; i < NPG; i += 256) {
    const float* xr = xn + (base + i) * DIM;
    float e = 0.0f;
    for (int k = 0; k < DIM; ++k) e += xr[k] * sh[k];
    sa[i] = e;
    lmax = fmaxf(lmax, e);
  }
  se[tid] = lmax; __syncthreads();
  for (int s = 128; s > 0; s >>= 1) { if (tid < s) se[tid] = fmaxf(se[tid], se[tid + s]); __syncthreads(); }
  float gmax = se[0]; __syncthreads();
  float lsum = 0.0f;
  for (int i = tid; i < NPG; i += 256) {
    float w = expf(sa[i] - gmax);
    sa[i] = w;
    lsum += w;
  }
  se[tid] = lsum; __syncthreads();
  for (int s = 128; s > 0; s >>= 1) { if (tid < s) se[tid] += se[tid + s]; __syncthreads(); }
  float gsum = se[0]; __syncthreads();
  if (tid < DIM) {
    float r = 0.0f;
    for (int i = 0; i < NPG; ++i) r += sa[i] * xn[(base + i) * DIM + tid];
    qstar[(size_t)g * 2 * DIM + tid]       = sh[tid];
    qstar[(size_t)g * 2 * DIM + DIM + tid] = r / gsum;
  }
}

// reparameterize (per-graph constant -> mean is identity) + fc1/relu + fc2
__global__ __launch_bounds__(128) void final_head(const float* __restrict__ qmu,
                                                  const float* __restrict__ qlv,
                                                  const float* __restrict__ eps,   // [G,128]
                                                  const float* __restrict__ fc1_w, // [128,64]
                                                  const float* __restrict__ fc1_b,
                                                  const float* __restrict__ fc2_w, // [64,1]
                                                  const float* __restrict__ fc2_b,
                                                  float* __restrict__ out) {
  int g = blockIdx.x, t = threadIdx.x;
  __shared__ float lat[2 * DIM];
  __shared__ float o1[DIM];
  __shared__ float red[DIM];
  lat[t] = eps[(size_t)g * 2 * DIM + t] * expf(0.5f * qlv[(size_t)g * 2 * DIM + t]) +
           qmu[(size_t)g * 2 * DIM + t];
  __syncthreads();
  if (t < DIM) {
    float acc = fc1_b[t];
    for (int j = 0; j < 2 * DIM; ++j) acc += lat[j] * fc1_w[(size_t)j * DIM + t];
    o1[t] = fmaxf(acc, 0.0f);
  }
  __syncthreads();
  if (t < DIM) red[t] = o1[t] * fc2_w[t];
  __syncthreads();
  for (int s = 32; s > 0; s >>= 1) { if (t < s) red[t] += red[t + s]; __syncthreads(); }
  if (t == 0) out[g] = red[0] + fc2_b[0];
}

// ------------------------------- launcher -----------------------------------
// Input order (setup_inputs dict order, params flattened depth-first in
// insertion order): x, edge_index, edge_attr, batch, eps_class,
// lin0_w, lin0_b, conv{w1,b1,w2,b2,bias}, gru{w_ih,w_hh,b_ih,b_hh},
// node_mu_conv{5}, node_lv_conv{5}, graph_mu_conv{5}, graph_lv_conv{5},
// s2s_mu{w_ih,w_hh,b_ih,b_hh}, s2s_lv{4}, fc1_w, fc1_b, fc2_w, fc2_b.
extern "C" void kernel_launch(void* const* d_in, const int* in_sizes, int n_in,
                              void* d_out, int out_size, void* d_ws, size_t ws_size,
                              hipStream_t stream) {
  (void)in_sizes; (void)n_in; (void)out_size; (void)ws_size;
  const float* x         = (const float*)d_in[0];
  const int*   ei        = (const int*)d_in[1];
  const int*   src       = ei;
  const int*   tgt       = ei + N_EDGES;
  const float* edge_attr = (const float*)d_in[2];
  const float* eps       = (const float*)d_in[4];
  const float* lin0_w = (const float*)d_in[5];
  const float* lin0_b = (const float*)d_in[6];
  const float* c_w1 = (const float*)d_in[7],  *c_b1 = (const float*)d_in[8];
  const float* c_w2 = (const float*)d_in[9],  *c_b2 = (const float*)d_in[10];
  const float* c_bias = (const float*)d_in[11];
  const float* g_wih = (const float*)d_in[12], *g_whh = (const float*)d_in[13];
  const float* g_bih = (const float*)d_in[14], *g_bhh = (const float*)d_in[15];
  // node_mu_conv (16..20) / node_lv_conv (21..25): dead in the reference — skipped.
  const float* gm_w1 = (const float*)d_in[26], *gm_b1 = (const float*)d_in[27];
  const float* gm_w2 = (const float*)d_in[28], *gm_b2 = (const float*)d_in[29];
  const float* gm_bias = (const float*)d_in[30];
  const float* gl_w1 = (const float*)d_in[31], *gl_b1 = (const float*)d_in[32];
  const float* gl_w2 = (const float*)d_in[33], *gl_b2 = (const float*)d_in[34];
  const float* gl_bias = (const float*)d_in[35];
  const float* mu_wih = (const float*)d_in[36], *mu_whh = (const float*)d_in[37];
  const float* mu_bih = (const float*)d_in[38], *mu_bhh = (const float*)d_in[39];
  const float* lv_wih = (const float*)d_in[40], *lv_whh = (const float*)d_in[41];
  const float* lv_bih = (const float*)d_in[42], *lv_bhh = (const float*)d_in[43];
  const float* fc1_w = (const float*)d_in[44], *fc1_b = (const float*)d_in[45];
  const float* fc2_w = (const float*)d_in[46], *fc2_b = (const float*)d_in[47];

  // ---- workspace carve-up (all 256B aligned, ~36 MB total) ----
  char* ws = (char*)d_ws;
  size_t off = 0;
  auto carve = [&](size_t bytes) -> void* {
    void* p = ws + off;
    off = (off + bytes + 255) & ~(size_t)255;
    return p;
  };
  const size_t node_feat_bytes = (size_t)N_NODES * DIM * sizeof(float);
  float*  accum = (float*)carve(node_feat_bytes);
  float*  hA    = (float*)carve(node_feat_bytes);
  float*  hB    = (float*)carve(node_feat_bytes);
  float*  gmu   = (float*)carve(node_feat_bytes);
  float*  glv   = (float*)carve(node_feat_bytes);
  float*  deg   = (float*)carve((size_t)N_NODES * sizeof(float));
  __bf16* Wt0 = (__bf16*)carve(WT_ELEMS * sizeof(__bf16));
  __bf16* Wt1 = (__bf16*)carve(WT_ELEMS * sizeof(__bf16));
  __bf16* Wt2 = (__bf16*)carve(WT_ELEMS * sizeof(__bf16));
  float*  qmu = (float*)carve((size_t)NGRAPH * 2 * DIM * sizeof(float));
  float*  qlv = (float*)carve((size_t)NGRAPH * 2 * DIM * sizeof(float));
  float*  hs  = (float*)carve((size_t)NGRAPH * DIM * sizeof(float));
  float*  cs  = (float*)carve((size_t)NGRAPH * DIM * sizeof(float));

  const int nd_blocks  = (N_NODES * DIM + 255) / 256;
  const int tile_cnt   = N_EDGES / 16;
  const int msg_blocks = (tile_cnt + NW - 1) / NW;
  const int wt_blocks  = (int)((WT_ELEMS + 255) / 256);

  // degree (edge_index fixed -> once)
  hipMemsetAsync(deg, 0, (size_t)N_NODES * sizeof(float), stream);
  deg_kernel<<<(N_EDGES + 255) / 256, 256, 0, stream>>>(tgt, deg, N_EDGES);

  // reorganize w2/b2 for the three live conv heads (1 MB each -> lives in L2)
  prep_w_kernel<<<wt_blocks, 256, 0, stream>>>(c_w2,  c_b2,  Wt0);
  prep_w_kernel<<<wt_blocks, 256, 0, stream>>>(gm_w2, gm_b2, Wt1);
  prep_w_kernel<<<wt_blocks, 256, 0, stream>>>(gl_w2, gl_b2, Wt2);

  // lin0 + relu
  lin0_kernel<<<nd_blocks, 256, 0, stream>>>(x, lin0_w, lin0_b, hA, N_NODES);

  // 3x shared NNConv + GRU (ping-pong hA/hB); cur==h==out throughout
  float* cur = hA;
  float* nxt = hB;
  for (int it = 0; it < 3; ++it) {
    hipMemsetAsync(accum, 0, node_feat_bytes, stream);
    edge_msg_wmma<<<msg_blocks, NW * 32, 0, stream>>>(cur, src, tgt, edge_attr,
                                                      c_w1, c_b1, Wt0, accum, N_EDGES);
    node_update<<<nd_blocks, 256, 0, stream>>>(accum, deg, c_bias, accum, N_NODES);
    gru_kernel<<<nd_blocks, 256, 0, stream>>>(accum, cur, g_wih, g_bih, g_whh, g_bhh,
                                              nxt, N_NODES);
    float* t = cur; cur = nxt; nxt = t;
  }
  // cur now holds the final node state ("out")

  // graph heads
  hipMemsetAsync(accum, 0, node_feat_bytes, stream);
  edge_msg_wmma<<<msg_blocks, NW * 32, 0, stream>>>(cur, src, tgt, edge_attr,
                                                    gm_w1, gm_b1, Wt1, accum, N_EDGES);
  node_update<<<nd_blocks, 256, 0, stream>>>(accum, deg, gm_bias, gmu, N_NODES);

  hipMemsetAsync(accum, 0, node_feat_bytes, stream);
  edge_msg_wmma<<<msg_blocks, NW * 32, 0, stream>>>(cur, src, tgt, edge_attr,
                                                    gl_w1, gl_b1, Wt2, accum, N_EDGES);
  node_update<<<nd_blocks, 256, 0, stream>>>(accum, deg, gl_bias, glv, N_NODES);

  // two set2sets (G=125, tiny)
  for (int which = 0; which < 2; ++which) {
    const float* xn  = which ? glv : gmu;
    float*       qs  = which ? qlv : qmu;
    const float* wih = which ? lv_wih : mu_wih;
    const float* whh = which ? lv_whh : mu_whh;
    const float* bih = which ? lv_bih : mu_bih;
    const float* bhh = which ? lv_bhh : mu_bhh;
    hipMemsetAsync(qs, 0, (size_t)NGRAPH * 2 * DIM * sizeof(float), stream);
    hipMemsetAsync(hs, 0, (size_t)NGRAPH * DIM * sizeof(float), stream);
    hipMemsetAsync(cs, 0, (size_t)NGRAPH * DIM * sizeof(float), stream);
    for (int s = 0; s < 3; ++s) {
      s2s_lstm<<<NGRAPH, 64, 0, stream>>>(qs, hs, cs, wih, bih, whh, bhh);
      s2s_attn<<<NGRAPH, 256, 0, stream>>>(xn, hs, qs);
    }
  }

  // reparameterize + fc1/relu + fc2
  final_head<<<NGRAPH, 128, 0, stream>>>(qmu, qlv, eps, fc1_w, fc1_b, fc2_w, fc2_b,
                                         (float*)d_out);
}
